// ScaledDotProductAttention_82815559402140
// MI455X (gfx1250) — compile-verified, matrix operand order checked by reading, not run on hardware
//
#include <hip/hip_runtime.h>

typedef __attribute__((ext_vector_type(16))) _Float16 v16h;
typedef __attribute__((ext_vector_type(8)))  float    v8f;
typedef __attribute__((ext_vector_type(4)))  float    v4f;

#define BATCH 2
#define HEADS 16
#define SEQ   2048
#define DIM   64
#define ROWP  (SEQ + 4)          // padded LDS row stride (floats); 16B-aligned rows
#define INV_TEMP 0.125f          // 1/sqrt(64)
#define NEG_BIG (-1.0e9f)

__global__ __launch_bounds__(256)
void ScaledDotProductAttention_82815559402140_kernel(
    const float* __restrict__ q, const float* __restrict__ k,
    const float* __restrict__ v, const int* __restrict__ mask,
    const float* __restrict__ prior, float* __restrict__ out,
    float* __restrict__ attn)
{
    __shared__ float sc[16 * ROWP];   // 16 query rows x 2048 keys (padded), fp32
    __shared__ float part[4 * 256];   // P@V partial reduction buffer

    const int tid  = threadIdx.x;
    const int lane = tid & 31;
    const int wave = tid >> 5;        // 0..7
    const int half = lane >> 4;       // 0/1
    const int l16  = lane & 15;

    const int qtile = blockIdx.x & 127;     // SEQ/16
    const int bh    = blockIdx.x >> 7;      // 0..B*H-1
    const int b     = bh >> 4;              // /HEADS
    const int q0    = qtile * 16;

    const float* qb = q + (size_t)bh * SEQ * DIM;
    const float* kb = k + (size_t)bh * SEQ * DIM;
    const float* vb = v + (size_t)bh * SEQ * DIM;
    const int*   mb = mask  + (size_t)b * SEQ;
    const float* pb = prior + (size_t)b * SEQ * SEQ;

    // ---------- Phase 1: persistent Q A-fragments (16x32 f16 layout, 2 chunks of K=32) ----------
    v16h aq0, aq1;
    {
        const float* qr = qb + (size_t)(q0 + l16) * DIM;
        const int base = 8 * half;
        #pragma unroll
        for (int e = 0; e < 8; ++e) {
            aq0[e]     = (_Float16)(qr[base + e]          * INV_TEMP);
            aq0[8 + e] = (_Float16)(qr[base + 16 + e]     * INV_TEMP);
            aq1[e]     = (_Float16)(qr[32 + base + e]     * INV_TEMP);
            aq1[8 + e] = (_Float16)(qr[32 + base + 16 + e]* INV_TEMP);
        }
    }

    // ---------- Phase 2: scores = (Q/T)K^T * prior, masked; 8 waves split 128 key-tiles ----------
    for (int i = 0; i < 16; ++i) {
        const int kt   = wave + 8 * i;
        const int krow = kt * 16 + l16;       // this lane's key row = B-matrix column N
        const float* kr = kb + (size_t)krow * DIM;
        const int base = 8 * half;
        v16h bk0, bk1;
        #pragma unroll
        for (int e = 0; e < 8; ++e) {
            bk0[e]     = (_Float16)kr[base + e];
            bk0[8 + e] = (_Float16)kr[base + 16 + e];
            bk1[e]     = (_Float16)kr[32 + base + e];
            bk1[8 + e] = (_Float16)kr[32 + base + 16 + e];
        }
        v8f c = {};
        c = __builtin_amdgcn_wmma_f32_16x16x32_f16(false, aq0, false, bk0, (short)0, c, false, false);
        c = __builtin_amdgcn_wmma_f32_16x16x32_f16(false, aq1, false, bk1, (short)0, c, false, false);

        const int col = kt * 16 + l16;        // C/D layout: lane -> N
        // Branchless mask: force unconditional (coalesced, batched) prior loads.
        const int   mv   = mb[col];
        const float mulf = (mv != 0) ? 1.0f : 0.0f;
        const float addf = (mv != 0) ? 0.0f : NEG_BIG;
        const float* prow = pb + (size_t)(q0 + 8 * half) * SEQ + col;
        float pw[8];
        #pragma unroll
        for (int r = 0; r < 8; ++r) pw[r] = prow[(size_t)r * SEQ];
        #pragma unroll
        for (int r = 0; r < 8; ++r) {
            const int m = 8 * half + r;       // C/D layout: vgpr r -> M = r + 8*half
            sc[m * ROWP + col] = fmaf(c[r] * pw[r], mulf, addf);
        }
    }
    __syncthreads();

    // ---------- Phase 3: softmax over keys (float4 passes); write attn to HBM (non-temporal) ----
    {
        const int r  = tid >> 4;              // row 0..15
        const int c0 = tid & 15;              // float4 lane within row
        v4f* row4 = (v4f*)(sc + r * ROWP);    // row base is 16B aligned (ROWP % 4 == 0)

        float mx = -3.4e38f;
        for (int i = 0; i < 32; ++i) {
            const v4f x = row4[c0 + 16 * i];
            mx = fmaxf(mx, fmaxf(fmaxf(x[0], x[1]), fmaxf(x[2], x[3])));
        }
        #pragma unroll
        for (int off = 1; off < 16; off <<= 1) mx = fmaxf(mx, __shfl_xor(mx, off, 16));

        float sum = 0.0f;
        for (int i = 0; i < 32; ++i) {
            v4f x = row4[c0 + 16 * i];
            v4f e;
            e[0] = __expf(x[0] - mx); e[1] = __expf(x[1] - mx);
            e[2] = __expf(x[2] - mx); e[3] = __expf(x[3] - mx);
            row4[c0 + 16 * i] = e;
            sum += (e[0] + e[1]) + (e[2] + e[3]);
        }
        #pragma unroll
        for (int off = 1; off < 16; off <<= 1) sum += __shfl_xor(sum, off, 16);
        const float inv = 1.0f / sum;

        v4f* arow4 = (v4f*)(attn + ((size_t)bh * SEQ + (q0 + r)) * SEQ);
        for (int i = 0; i < 32; ++i) {
            v4f p = row4[c0 + 16 * i];
            p[0] *= inv; p[1] *= inv; p[2] *= inv; p[3] *= inv;
            row4[c0 + 16 * i] = p;                       // keep probabilities for P@V
            __builtin_nontemporal_store(p, &arow4[c0 + 16 * i]);  // 512MB streaming write
        }
    }
    __syncthreads();

    // ---------- Phase 4: O = P @ V; waves split 4 d-tiles x 2 key-halves ----------
    {
        const int ntile = wave & 3;           // which 16-wide slice of D
        const int khalf = wave >> 2;          // which 1024-key half
        const int dcol  = ntile * 16 + l16;   // B/C/D layout: lane -> N (= output d)
        const int base  = 8 * half;
        v8f acc = {};
        for (int i = 0; i < 32; ++i) {
            const int kc = khalf * 1024 + i * 32;
            v16h pa, bv;
            const float* prow = sc + l16 * ROWP + kc + base;   // A: lane -> M row of P
            const float* vcol = vb + (size_t)(kc + base) * DIM + dcol;
            #pragma unroll
            for (int e = 0; e < 8; ++e) {
                pa[e]     = (_Float16)prow[e];
                pa[8 + e] = (_Float16)prow[16 + e];
                bv[e]     = (_Float16)vcol[(size_t)e * DIM];
                bv[8 + e] = (_Float16)vcol[(size_t)(16 + e) * DIM];
            }
            acc = __builtin_amdgcn_wmma_f32_16x16x32_f16(false, pa, false, bv, (short)0, acc, false, false);
        }
        if (khalf == 1) {
            #pragma unroll
            for (int r = 0; r < 8; ++r) part[ntile * 256 + r * 32 + lane] = acc[r];
        }
        __syncthreads();
        if (khalf == 0) {
            #pragma unroll
            for (int r = 0; r < 8; ++r) {
                const float val = acc[r] + part[ntile * 256 + r * 32 + lane];
                const int m = 8 * half + r;
                out[(size_t)bh * SEQ * DIM + (size_t)(q0 + m) * DIM + dcol] = val;
            }
        }
    }
}

extern "C" void kernel_launch(void* const* d_in, const int* in_sizes, int n_in,
                              void* d_out, int out_size, void* d_ws, size_t ws_size,
                              hipStream_t stream) {
    (void)in_sizes; (void)n_in; (void)out_size; (void)d_ws; (void)ws_size;
    const float* q     = (const float*)d_in[0];
    const float* k     = (const float*)d_in[1];
    const float* v     = (const float*)d_in[2];
    const int*   mask  = (const int*)d_in[3];
    const float* prior = (const float*)d_in[4];
    float* out  = (float*)d_out;
    float* attn = out + (size_t)BATCH * HEADS * SEQ * DIM;   // tuple order: (output, attn)

    const int nblocks = BATCH * HEADS * (SEQ / 16);          // 4096
    ScaledDotProductAttention_82815559402140_kernel<<<dim3(nblocks), dim3(256), 0, stream>>>(
        q, k, v, mask, prior, out, attn);
}
